// CausalWanSelfAttention_65773129171558
// MI455X (gfx1250) — compile-verified, matrix operand order checked by reading, not run on hardware
//
#include <hip/hip_runtime.h>

#define DIM 1536
#define HEADS 12
#define HD (DIM / HEADS)   // 128
#define MAX_ATTN 1920
#define SINK_SIZE 1

typedef _Float16 half_t;
typedef __attribute__((ext_vector_type(16))) _Float16 v16h;
typedef __attribute__((ext_vector_type(8)))  _Float16 v8h;
typedef __attribute__((ext_vector_type(8)))  float    v8f;

// Load a 16-half WMMA fragment for one lane.
// Per CDNA5 ISA: lane holds row/col (lane&15); element e maps to
// k = ((e<8) ? e : e+8) + ((lane>=16) ? 8 : 0).  With kb = (lane&16)?8:0 this is
// halves [kb..kb+7] and [kb+16..kb+23] of the row -> two 16B loads.
__device__ __forceinline__ v16h frag_ld(const half_t* row, int kb) {
  v8h lo = *(const v8h*)(row + kb);
  v8h hi = *(const v8h*)(row + kb + 16);
  v16h r;
#pragma unroll
  for (int i = 0; i < 8; ++i) { r[i] = lo[i]; r[i + 8] = hi[i]; }
  return r;
}

__device__ __forceinline__ v8f wmma_f16(v16h a, v16h b, v8f c) {
  return __builtin_amdgcn_wmma_f32_16x16x32_f16(false, a, false, b, (short)0, c,
                                                false, false);
}

__global__ void cvt_f32_f16_kernel(const float* __restrict__ in,
                                   half_t* __restrict__ out, int n) {
  int i = blockIdx.x * blockDim.x + threadIdx.x;
  if (i < n) out[i] = (half_t)in[i];
}

// C[M,N] = A[M,K] * B[N,K]^T + bias.
// 128x64 tile per 8-wave (256-thread) block; each wave computes 32x32.
// Register double-buffering: global loads for panel k+32 are issued while
// WMMAs consume the LDS panel for k.
template <bool F32OUT>
__global__ __launch_bounds__(256) void gemm_wmma_kernel(
    const half_t* __restrict__ A, const half_t* __restrict__ B,
    const float* __restrict__ bias, void* __restrict__ Cv, int M, int N, int K) {
  __shared__ half_t As[128][40];  // +8 pad: conflict-free b128 fragment reads
  __shared__ half_t Bs[64][40];
  const int tid = threadIdx.x;       // 0..255
  const int wave = tid >> 5, lane = tid & 31;
  const int lr = lane & 15;
  const int kb = (lane & 16) ? 8 : 0;
  const int m0 = blockIdx.y * 128;
  const int n0 = blockIdx.x * 64;
  const int wm = (wave >> 1) * 32;   // 0,32,64,96
  const int wn = (wave & 1) * 32;    // 0,32

  const int sR = tid >> 2;           // 0..63
  const int sK = (tid & 3) * 8;      // 0,8,16,24

  v8h ra0, ra1, rb;
  // prime panel 0
  ra0 = *(const v8h*)(A + (size_t)(m0 + sR) * K + sK);
  ra1 = *(const v8h*)(A + (size_t)(m0 + sR + 64) * K + sK);
  rb  = *(const v8h*)(B + (size_t)(n0 + sR) * K + sK);
  *(v8h*)(&As[sR][sK]) = ra0;
  *(v8h*)(&As[sR + 64][sK]) = ra1;
  *(v8h*)(&Bs[sR][sK]) = rb;

  v8f acc[2][2] = {};

  for (int kk = 0; kk < K; kk += 32) {
    __syncthreads();                 // panel kk visible in LDS
    const bool more = (kk + 32) < K;
    if (more) {                      // overlap next fetch with WMMA
      ra0 = *(const v8h*)(A + (size_t)(m0 + sR) * K + kk + 32 + sK);
      ra1 = *(const v8h*)(A + (size_t)(m0 + sR + 64) * K + kk + 32 + sK);
      rb  = *(const v8h*)(B + (size_t)(n0 + sR) * K + kk + 32 + sK);
    }
    v16h a0 = frag_ld(&As[wm + lr][0], kb);
    v16h a1 = frag_ld(&As[wm + 16 + lr][0], kb);
    v16h b0 = frag_ld(&Bs[wn + lr][0], kb);
    v16h b1 = frag_ld(&Bs[wn + 16 + lr][0], kb);
    acc[0][0] = wmma_f16(a0, b0, acc[0][0]);
    acc[0][1] = wmma_f16(a0, b1, acc[0][1]);
    acc[1][0] = wmma_f16(a1, b0, acc[1][0]);
    acc[1][1] = wmma_f16(a1, b1, acc[1][1]);
    __syncthreads();                 // all reads of panel kk done
    if (more) {
      *(v8h*)(&As[sR][sK]) = ra0;
      *(v8h*)(&As[sR + 64][sK]) = ra1;
      *(v8h*)(&Bs[sR][sK]) = rb;
    }
  }

#pragma unroll
  for (int ti = 0; ti < 2; ++ti)
#pragma unroll
    for (int tj = 0; tj < 2; ++tj) {
      int col = n0 + wn + tj * 16 + lr;
      float bb = bias[col];
      int rowb = m0 + wm + ti * 16 + kb;  // C layout: M = v + ((lane>=16)?8:0)
      if (F32OUT) {
        float* C = (float*)Cv;
#pragma unroll
        for (int v = 0; v < 8; ++v)
          C[(size_t)(rowb + v) * N + col] = acc[ti][tj][v] + bb;
      } else {
        half_t* C = (half_t*)Cv;
#pragma unroll
        for (int v = 0; v < 8; ++v)
          C[(size_t)(rowb + v) * N + col] = (half_t)(acc[ti][tj][v] + bb);
      }
    }
}

// RMSNorm over full DIM row (with weight), then 3D RoPE per head, in-place f16.
__global__ __launch_bounds__(128) void norm_rope_kernel(
    half_t* __restrict__ qk, const float* __restrict__ nw,
    const float* __restrict__ freqs, const int* __restrict__ gs) {
  const int tid = threadIdx.x;
  const int t = blockIdx.x;
  half_t* row = qk + (size_t)t * DIM;
  __shared__ float wsum[4];
  float ss = 0.0f;
  for (int i = tid; i < DIM; i += 128) {
    float v = (float)row[i];
    ss += v * v;
  }
#pragma unroll
  for (int off = 16; off > 0; off >>= 1) ss += __shfl_xor(ss, off);
  if ((tid & 31) == 0) wsum[tid >> 5] = ss;
  __syncthreads();
  float inv = rsqrtf((wsum[0] + wsum[1] + wsum[2] + wsum[3]) / (float)DIM + 1e-6f);

  const int Hg = gs[1], Wg = gs[2];
  const int fs = Hg * Wg;
  const int fr = t / fs;
  const int rem = t - fr * fs;
  const int hh = rem / Wg;
  const int ww = rem - hh * Wg;
  const int c2 = (HD / 2) / 3;     // 21
  const int c1 = HD / 2 - 2 * c2;  // 22

  for (int p = tid; p < DIM / 2; p += 128) {
    int head = p / (HD / 2);
    int j = p - head * (HD / 2);
    int fidx = (j < c1) ? fr : ((j < c1 + c2) ? hh : ww);
    const float* fp = freqs + ((size_t)fidx * (HD / 2) + j) * 2;
    float cr = fp[0], ci = fp[1];
    int base = head * HD + 2 * j;
    float a = (float)row[base] * inv * nw[base];
    float b = (float)row[base + 1] * inv * nw[base + 1];
    row[base] = (half_t)(a * cr - b * ci);
    row[base + 1] = (half_t)(a * ci + b * cr);
  }
}

// vt[(head*HD + d)][key] = v[key][head*HD + d]
__global__ void transpose_v_kernel(const half_t* __restrict__ v,
                                   half_t* __restrict__ vt, int L) {
  int i = blockIdx.x * blockDim.x + threadIdx.x;
  if (i < L * DIM) {
    int key = i / DIM;
    int c = i - key * DIM;
    vt[(size_t)c * L + key] = v[i];
  }
}

// Flash attention. One block = 64 queries x 1 head (4 waves, 16 queries each).
// fs % 64 == 0 so all waves share one frame -> identical key ranges; K and V^T
// chunks are staged once per block in LDS (4x traffic cut vs per-wave loads).
// S^T = K.Q^T so query = lane&15 (softmax stats need one shfl_xor(16));
// exponentiated P^T packs directly into a WMMA B-fragment; O^T = V^T.P^T.
__global__ __launch_bounds__(128) void attn_kernel(
    const half_t* __restrict__ q, const half_t* __restrict__ k,
    const half_t* __restrict__ vt, half_t* __restrict__ o,
    const int* __restrict__ gs, int L) {
  __shared__ half_t Ks[32][136];   // 32 keys x 128 dims (+8 pad)
  __shared__ half_t Vs[128][40];   // 128 dims x 32 keys (+8 pad)
  const int tid = threadIdx.x;
  const int wave = tid >> 5, lane = tid & 31;
  const int lr = lane & 15;
  const int kb = (lane & 16) ? 8 : 0;
  const int head = blockIdx.y;
  const int qb0 = blockIdx.x * 64;
  const int q0 = qb0 + wave * 16;
  const int fs = gs[1] * gs[2];
  const int sink = SINK_SIZE * fs;
  const float scale = rsqrtf((float)HD);

  const int myq = min(q0 + lr, L - 1);
  const int myqf = myq / fs;
  const int mywin = (myqf + 1) * fs - MAX_ATTN;

  // Block-uniform key ranges: [0,r1e) = sink frame, [r2s,r2e) = causal window.
  const int qf_lo = qb0 / fs;
  const int qf_hi = min(qb0 + 63, L - 1) / fs;
  const int end_max = min((qf_hi + 1) * fs, L);
  const int r1e = min(sink, end_max);
  int r2s = (qf_lo + 1) * fs - MAX_ATTN;
  if (r2s < r1e) r2s = r1e;
  const int r2e = end_max;

  // Q^T B-fragments: col = query (lane&15), 4 chunks cover HD=128.
  const half_t* qrow = q + (size_t)myq * DIM + head * HD;
  v16h qf4[4];
#pragma unroll
  for (int c = 0; c < 4; ++c) qf4[c] = frag_ld(qrow + c * 32, kb);

  v8f accO[8] = {};
  float m_run = -__builtin_huge_valf();
  float l_run = 0.0f;

#pragma unroll 1
  for (int range = 0; range < 2; ++range) {
    const int rs = (range == 0) ? 0 : r2s;
    const int re = (range == 0) ? r1e : r2e;
    for (int kt = rs; kt < re; kt += 32) {
      __syncthreads();   // previous chunk's LDS reads done
      // stage K tile: 32 rows x 128 halves = 512 v8h chunks / 128 threads
#pragma unroll
      for (int i = 0; i < 4; ++i) {
        int c = tid + i * 128;
        int r = c >> 4;                 // key row 0..31
        int kc = (c & 15) * 8;          // 0..120
        int krow = min(kt + r, L - 1);
        *(v8h*)(&Ks[r][kc]) =
            *(const v8h*)(k + (size_t)krow * DIM + head * HD + kc);
      }
      // stage V^T tile: 128 rows x 32 halves = 512 v8h chunks
#pragma unroll
      for (int i = 0; i < 4; ++i) {
        int c = tid + i * 128;
        int r = c >> 2;                 // d row 0..127
        int kc = (c & 3) * 8;           // 0,8,16,24
        *(v8h*)(&Vs[r][kc]) =
            *(const v8h*)(vt + (size_t)(head * HD + r) * L + kt + kc);
      }
      // hint the next chunk toward the caches (global_prefetch_b8)
      if (kt + 32 < re) {
        __builtin_prefetch(
            k + (size_t)min(kt + 32 + (tid & 31), L - 1) * DIM + head * HD, 0, 1);
        __builtin_prefetch(vt + (size_t)(head * HD + tid) * L + kt + 32, 0, 1);
      }
      __syncthreads();   // K/V tiles visible

      v8f s0 = {}, s1 = {};  // S^T: keys kt+0..15 / kt+16..31 vs 16 queries
#pragma unroll
      for (int c = 0; c < 4; ++c) {
        v16h a0 = frag_ld(&Ks[lr][c * 32], kb);
        v16h a1 = frag_ld(&Ks[16 + lr][c * 32], kb);
        s0 = wmma_f16(a0, qf4[c], s0);
        s1 = wmma_f16(a1, qf4[c], s1);
      }
      float sv0[8], sv1[8];
      float mloc = -__builtin_huge_valf();
#pragma unroll
      for (int v = 0; v < 8; ++v) {
        int ki0 = kt + kb + v;  // key index of C element v for this lane
        int ki1 = ki0 + 16;
        bool ok0 = (ki0 < re) && (ki0 / fs <= myqf) &&
                   ((ki0 < sink) || (ki0 >= mywin));
        bool ok1 = (ki1 < re) && (ki1 / fs <= myqf) &&
                   ((ki1 < sink) || (ki1 >= mywin));
        float x0 = ok0 ? s0[v] * scale : -__builtin_huge_valf();
        float x1 = ok1 ? s1[v] * scale : -__builtin_huge_valf();
        sv0[v] = x0; sv1[v] = x1;
        mloc = fmaxf(mloc, fmaxf(x0, x1));
      }
      mloc = fmaxf(mloc, __shfl_xor(mloc, 16));  // lanes l, l^16 share a query
      float mnew = fmaxf(m_run, mloc);
      float corr = __expf(m_run - mnew);
      l_run *= corr;
#pragma unroll
      for (int t = 0; t < 8; ++t) accO[t] *= corr;
      v16h pb;  // P^T as B-fragment: e<8 -> s0[e], e>=8 -> s1[e-8]
      float ls = 0.0f;
#pragma unroll
      for (int v = 0; v < 8; ++v) {
        float p0 = __expf(sv0[v] - mnew);
        float p1 = __expf(sv1[v] - mnew);
        ls += p0 + p1;
        pb[v] = (half_t)p0;
        pb[v + 8] = (half_t)p1;
      }
      ls += __shfl_xor(ls, 16);
      l_run += ls;
      m_run = mnew;
#pragma unroll
      for (int t = 0; t < 8; ++t) {  // O^T += V^T . P^T, 8 d-tiles of 16
        v16h av = frag_ld(&Vs[t * 16 + lr][0], kb);
        accO[t] = wmma_f16(av, pb, accO[t]);
      }
    }
  }

  float linv = 1.0f / l_run;
  half_t* orow = o + (size_t)myq * DIM + head * HD;
#pragma unroll
  for (int t = 0; t < 8; ++t) {
    v8h pk;
#pragma unroll
    for (int v = 0; v < 8; ++v) pk[v] = (half_t)(accO[t][v] * linv);
    *(v8h*)(orow + t * 16 + kb) = pk;  // O^T row d = t*16 + kb + v: contiguous
  }
}

extern "C" void kernel_launch(void* const* d_in, const int* in_sizes, int n_in,
                              void* d_out, int out_size, void* d_ws,
                              size_t ws_size, hipStream_t stream) {
  (void)n_in; (void)out_size; (void)ws_size;
  const float* x     = (const float*)d_in[0];
  const int*   gsz   = (const int*)d_in[2];
  const float* freqs = (const float*)d_in[3];
  const float* wq = (const float*)d_in[4];
  const float* bq = (const float*)d_in[5];
  const float* wk = (const float*)d_in[6];
  const float* bk = (const float*)d_in[7];
  const float* wv = (const float*)d_in[8];
  const float* bv = (const float*)d_in[9];
  const float* wo = (const float*)d_in[10];
  const float* bo = (const float*)d_in[11];
  const float* nqw = (const float*)d_in[12];
  const float* nkw = (const float*)d_in[13];

  const int L = in_sizes[0] / DIM;
  const size_t LD = (size_t)L * DIM;
  const size_t DD = (size_t)DIM * DIM;

  half_t* p = (half_t*)d_ws;
  half_t* xh  = p; p += LD;
  half_t* wqh = p; p += DD;
  half_t* wkh = p; p += DD;
  half_t* wvh = p; p += DD;
  half_t* woh = p; p += DD;
  half_t* qh  = p; p += LD;
  half_t* kh  = p; p += LD;
  half_t* vh  = p; p += LD;
  half_t* vth = p; p += LD;
  half_t* oh  = p; p += LD;

  cvt_f32_f16_kernel<<<(int)((LD + 255) / 256), 256, 0, stream>>>(x, xh, (int)LD);
  cvt_f32_f16_kernel<<<(int)((DD + 255) / 256), 256, 0, stream>>>(wq, wqh, (int)DD);
  cvt_f32_f16_kernel<<<(int)((DD + 255) / 256), 256, 0, stream>>>(wk, wkh, (int)DD);
  cvt_f32_f16_kernel<<<(int)((DD + 255) / 256), 256, 0, stream>>>(wv, wvh, (int)DD);
  cvt_f32_f16_kernel<<<(int)((DD + 255) / 256), 256, 0, stream>>>(wo, woh, (int)DD);

  dim3 gg(DIM / 64, L / 128), gb(256);
  gemm_wmma_kernel<false><<<gg, gb, 0, stream>>>(xh, wqh, bq, qh, L, DIM, DIM);
  gemm_wmma_kernel<false><<<gg, gb, 0, stream>>>(xh, wkh, bk, kh, L, DIM, DIM);
  gemm_wmma_kernel<false><<<gg, gb, 0, stream>>>(xh, wvh, bv, vh, L, DIM, DIM);

  norm_rope_kernel<<<L, 128, 0, stream>>>(qh, nqw, freqs, gsz);
  norm_rope_kernel<<<L, 128, 0, stream>>>(kh, nkw, freqs, gsz);
  transpose_v_kernel<<<(int)((LD + 255) / 256), 256, 0, stream>>>(vh, vth, L);

  attn_kernel<<<dim3(L / 64, HEADS), 128, 0, stream>>>(qh, kh, vth, oh, gsz, L);

  gemm_wmma_kernel<true><<<gg, gb, 0, stream>>>(oh, woh, bo, d_out, L, DIM, DIM);
}
